// gnn_classifier_6408091206384
// MI455X (gfx1250) — matrix operand order repaired and checked by reading, not compiled, not run on hardware
//
#include <hip/hip_runtime.h>
#include <hip/hip_bf16.h>

// ---------------------------------------------------------------------------
// CDNA5 (gfx1250) implementation: BiLSTM + 12x Mamba + attention head.
// All dense GEMMs run through a bf16 WMMA (v_wmma_f32_16x16x32_bf16) kernel
// with async global->LDS tile staging (ASYNCcnt) when the toolchain exposes
// the builtin, falling back to global_load+ds_store otherwise.
// Sequential parts (LSTM recurrence, selective scan) are VALU kernels.
// Workspace layout (~97MB) lives in d_ws.
// ---------------------------------------------------------------------------

typedef __attribute__((ext_vector_type(16))) __bf16 bf16x16_t;
typedef __attribute__((ext_vector_type(8)))  float  f32x8;
typedef __attribute__((ext_vector_type(4)))  unsigned int u32x4;

union V16 { bf16x16_t v; u32x4 u[2]; };

#if defined(__gfx1250__) && __has_builtin(__builtin_amdgcn_global_load_async_to_lds_b128)
#define ASYNC_LDS 1
#else
#define ASYNC_LDS 0
#endif

#if ASYNC_LDS
// Builtin signature (from compiler diagnostic): param 1 is a pointer to a
// 16-byte int vector in the global (device) address space; param 2 the LDS
// counterpart; then immediate offset + cpol.
typedef __attribute__((__vector_size__(4 * sizeof(int)))) int v4i_vs;
typedef __attribute__((address_space(1))) v4i_vs v4i_g;
typedef __attribute__((address_space(3))) v4i_vs v4i_l;

// 16-byte async global->LDS copy; OFF is the shared instruction immediate
// offset (applied to both the global and LDS addresses per the ISA).
template <int OFF>
__device__ __forceinline__ void async_copy16(const void* g, void* l) {
  __builtin_amdgcn_global_load_async_to_lds_b128(
      (v4i_g*)(uintptr_t)g,
      (v4i_l*)(unsigned)(uintptr_t)l,
      OFF, 0);
}
#endif

__device__ __forceinline__ void async_wait_all() {
#if defined(__gfx1250__)
#if __has_builtin(__builtin_amdgcn_s_wait_asynccnt)
  __builtin_amdgcn_s_wait_asynccnt(0);
#else
  asm volatile("s_wait_asynccnt 0x0" ::: "memory");
#endif
#endif
}

__device__ __forceinline__ __bf16 f2bf(float f) {
  unsigned u = __builtin_bit_cast(unsigned, f);
  unsigned r = (u + 0x7FFFu + ((u >> 16) & 1u)) >> 16;
  unsigned short h = (unsigned short)r;
  return __builtin_bit_cast(__bf16, h);
}

__device__ __forceinline__ float sigf(float x) { return 1.0f / (1.0f + __expf(-x)); }

// ---------------------------------------------------------------------------
// Tiled GEMM:  C[M x N] = A[M x K] * W[N x K]^T   (bf16 inputs, f32 accum)
// Block: 128 threads (4 waves), tile 64x64, K step 64 (2 WMMA K-slabs/stage).
// Fragment layouts per CDNA5 ISA 7.12.2 (wave32).
// ---------------------------------------------------------------------------
__global__ __launch_bounds__(128) void gemm_bf16_kernel(
    const __bf16* __restrict__ A, const __bf16* __restrict__ W,
    float* __restrict__ C, int M, int N, int K, int ldc)
{
  __shared__ __attribute__((aligned(16))) __bf16 As[64 * 64];
  __shared__ __attribute__((aligned(16))) __bf16 Ws[64 * 64];

  const int tid  = threadIdx.x;
  const int wave = tid >> 5;
  const int lane = tid & 31;
  const int m0 = blockIdx.x * 64;
  const int n0 = blockIdx.y * 64;
  const int wm = (wave & 1) * 32;
  const int wn = (wave >> 1) * 32;

  f32x8 zero8 = {0.f,0.f,0.f,0.f,0.f,0.f,0.f,0.f};
  f32x8 acc[2][2];
  acc[0][0] = zero8; acc[0][1] = zero8; acc[1][0] = zero8; acc[1][1] = zero8;

  const int r  = tid >> 1;             // 0..63 (tile row)
  const int hh = (tid & 1) * 32;       // element offset within 64-wide K slab
  const bool wok = (n0 + r) < N;

  const int mlane = lane & 15;
  const int ka = (lane < 16) ? 0 : 8;  // A: K {0-7,16-23} | {8-15,24-31}
  const int kb = (lane < 16) ? 0 : 8;  // B: K {0-7,16-23} | {8-15,24-31}

  for (int k0 = 0; k0 < K; k0 += 64) {
    const __bf16* ga = A + (size_t)(m0 + r) * K + k0 + hh;
    const __bf16* gw = W + (size_t)(n0 + r) * K + k0 + hh;
    __bf16* la = As + r * 64 + hh;
    __bf16* lw = Ws + r * 64 + hh;

#if ASYNC_LDS
    async_copy16<0>(ga, la);
    async_copy16<16>(ga, la);
    if (wok) {
      async_copy16<0>(gw, lw);
      async_copy16<16>(gw, lw);
    } else {
      u32x4 z = {0u, 0u, 0u, 0u};
      *(u32x4*)lw = z;
      *(u32x4*)(lw + 8) = z;
    }
    if (k0 + 64 < K) {                  // warm next slab while WMMAs run
      __builtin_prefetch(ga + 64, 0, 3);
      if (wok) __builtin_prefetch(gw + 64, 0, 3);
    }
    async_wait_all();
#else
    {
      const u32x4* src = (const u32x4*)ga;
      u32x4* dst = (u32x4*)la;
      dst[0] = src[0];
      dst[1] = src[1];
    }
    {
      u32x4* dst = (u32x4*)lw;
      if (wok) {
        const u32x4* src = (const u32x4*)gw;
        dst[0] = src[0];
        dst[1] = src[1];
      } else {
        u32x4 z = {0u, 0u, 0u, 0u};
        dst[0] = z; dst[1] = z;
      }
    }
#endif
    __syncthreads();

#pragma unroll
    for (int ks = 0; ks < 2; ++ks) {
      const int kbase = ks * 32;
      V16 af[2], bfr[2];
#pragma unroll
      for (int mi = 0; mi < 2; ++mi) {
        const __bf16* base = As + (wm + mi * 16 + mlane) * 64 + kbase;
        af[mi].u[0] = *(const u32x4*)(base + ka);
        af[mi].u[1] = *(const u32x4*)(base + 16 + ka);
      }
#pragma unroll
      for (int ni = 0; ni < 2; ++ni) {
        const __bf16* base = Ws + (wn + ni * 16 + mlane) * 64 + kbase;
        bfr[ni].u[0] = *(const u32x4*)(base + kb);
        bfr[ni].u[1] = *(const u32x4*)(base + kb + 16);
      }
#pragma unroll
      for (int mi = 0; mi < 2; ++mi)
#pragma unroll
        for (int ni = 0; ni < 2; ++ni)
          acc[mi][ni] = __builtin_amdgcn_wmma_f32_16x16x32_bf16(
              false, af[mi].v, false, bfr[ni].v, (short)0, acc[mi][ni],
              false, false);
    }
    __syncthreads();
  }

  // --- store: VGPR v -> M = v + (lane<16 ? 0 : 8), N = lane%16 ---
  const int nlane = lane & 15;
  const int mbase = (lane < 16) ? 0 : 8;
#pragma unroll
  for (int mi = 0; mi < 2; ++mi)
#pragma unroll
    for (int ni = 0; ni < 2; ++ni) {
      int gn = n0 + wn + ni * 16 + nlane;
      if (gn < N) {
#pragma unroll
        for (int v = 0; v < 8; ++v) {
          int gm = m0 + wm + mi * 16 + mbase + v;
          C[(size_t)gm * ldc + gn] = acc[mi][ni][v];
        }
      }
    }
}

// ---------------------------------------------------------------------------
// Small conversion / staging kernels
// ---------------------------------------------------------------------------
__global__ __launch_bounds__(256) void cvt_bf16_kernel(const float* __restrict__ in,
                                                       __bf16* __restrict__ out, int n)
{
  int i = blockIdx.x * 256 + threadIdx.x;
  if (i < n) out[i] = f2bf(in[i]);
}

// dt_w (1280x40) -> bf16 padded to 1280x64 (K padded with zeros)
__global__ __launch_bounds__(256) void cvt_dtw_kernel(const float* __restrict__ w,
                                                      __bf16* __restrict__ out)
{
  int i = blockIdx.x * 256 + threadIdx.x;
  if (i < 1280 * 64) {
    int nrow = i >> 6, k = i & 63;
    out[i] = (k < 40) ? f2bf(w[nrow * 40 + k]) : f2bf(0.f);
  }
}

// dbc[:, :40] -> bf16 padded to 2048x64
__global__ __launch_bounds__(256) void cvt_dt_kernel(const float* __restrict__ dbc,
                                                     __bf16* __restrict__ out)
{
  int i = blockIdx.x * 256 + threadIdx.x;
  if (i < 2048 * 64) {
    int r2 = i >> 6, k = i & 63;
    out[i] = (k < 40) ? f2bf(dbc[(size_t)r2 * 80 + k]) : f2bf(0.f);
  }
}

// gather embedding rows into padded [8*256 x 1280] bf16 (zero pad)
__global__ __launch_bounds__(256) void gather_emb_kernel(
    const float* __restrict__ emb, const long long* __restrict__ seg,
    __bf16* __restrict__ out)
{
  int row = blockIdx.x;                       // 0..2047
  int c = blockIdx.y * 256 + threadIdx.x;     // 0..1279
  int s = row >> 8, t = row & 255;
  long long st = seg[2 * s];
  int L = (int)(seg[2 * s + 1] - st + 1);
  float v = (t < L) ? emb[(size_t)(st + t) * 1280 + c] : 0.f;
  out[(size_t)row * 1280 + c] = f2bf(v);
}

__global__ __launch_bounds__(256) void add_kernel(float* __restrict__ x,
                                                  const float* __restrict__ y, int n)
{
  int i = blockIdx.x * 256 + threadIdx.x;
  if (i < n) x[i] += y[i];
}

// ---------------------------------------------------------------------------
// BiLSTM recurrence: one workgroup per (segment, direction)
// ---------------------------------------------------------------------------
__global__ __launch_bounds__(256) void lstm_kernel(
    const float* __restrict__ pre_f, const float* __restrict__ pre_b,
    const float* __restrict__ Whh_f, const float* __restrict__ Whh_b,
    const float* __restrict__ b_f, const float* __restrict__ b_b,
    const long long* __restrict__ seg,
    float* __restrict__ te, float* __restrict__ x)
{
  int blk = blockIdx.x;              // 0..15
  int s = blk >> 1, dir = blk & 1;
  int L = (int)(seg[2 * s + 1] - seg[2 * s] + 1);
  const float* pre  = dir ? pre_b : pre_f;
  const float* Whh  = dir ? Whh_b : Whh_f;
  const float* bias = dir ? b_b : b_f;

  __shared__ float hsh[320], csh[320], gsh[1280];
  int tid = threadIdx.x;
  for (int j = tid; j < 320; j += 256) { hsh[j] = 0.f; csh[j] = 0.f; }
  __syncthreads();

  for (int step = 0; step < L; ++step) {
    int t = dir ? (L - 1 - step) : step;
    const float* prow = pre + (size_t)(s * 256 + t) * 1280;

    for (int j = tid; j < 1280; j += 256) {
      const float* w = Whh + (size_t)j * 320;
      float acc = prow[j] + bias[j];
#pragma unroll 4
      for (int k = 0; k < 320; k += 4) {
        acc += w[k] * hsh[k] + w[k + 1] * hsh[k + 1] +
               w[k + 2] * hsh[k + 2] + w[k + 3] * hsh[k + 3];
      }
      gsh[j] = acc;
    }
    __syncthreads();

    for (int j = tid; j < 320; j += 256) {
      float i_ = sigf(gsh[j]);
      float f_ = sigf(gsh[320 + j]);
      float g_ = tanhf(gsh[640 + j]);
      float o_ = sigf(gsh[960 + j]);
      float cc = f_ * csh[j] + i_ * g_;
      float hv = o_ * tanhf(cc);
      csh[j] = cc;
      hsh[j] = hv;
      size_t row = (size_t)(s * 256 + t);
      te[row * 640 + dir * 320 + j] = hv;
      x [row * 640 + dir * 320 + j] = hv;
    }
    __syncthreads();
  }
}

// ---------------------------------------------------------------------------
// RMSNorm -> bf16 (rows of 640)
// ---------------------------------------------------------------------------
__global__ __launch_bounds__(256) void rmsnorm_bf16_kernel(
    const float* __restrict__ x, const float* __restrict__ w,
    __bf16* __restrict__ out)
{
  int row = blockIdx.x, tid = threadIdx.x;
  __shared__ float red[256];
  const float* xr = x + (size_t)row * 640;
  float p = 0.f;
  for (int c = tid; c < 640; c += 256) { float v = xr[c]; p += v * v; }
  red[tid] = p; __syncthreads();
  for (int o = 128; o > 0; o >>= 1) {
    if (tid < o) red[tid] += red[tid + o];
    __syncthreads();
  }
  float rs = rsqrtf(red[0] / 640.f + 1e-5f);
  for (int c = tid; c < 640; c += 256)
    out[(size_t)row * 640 + c] = f2bf(xr[c] * rs * w[c]);
}

// ---------------------------------------------------------------------------
// Causal depthwise conv (D_CONV=4) + SiLU ; writes f32 xs and bf16 staging
// ---------------------------------------------------------------------------
__global__ __launch_bounds__(256) void conv_silu_kernel(
    const float* __restrict__ xz, const float* __restrict__ conv_w,
    const float* __restrict__ conv_b, const long long* __restrict__ seg,
    float* __restrict__ xs, __bf16* __restrict__ xs_bf)
{
  int row = blockIdx.x;                      // 0..2047
  int e = blockIdx.y * 256 + threadIdx.x;    // 0..1279
  int s = row >> 8, t = row & 255;
  int L = (int)(seg[2 * s + 1] - seg[2 * s] + 1);
  float v = 0.f;
  if (t < L) {
    v = conv_b[e];
#pragma unroll
    for (int k = 0; k < 4; ++k) {
      int tt = t + k - 3;
      if (tt >= 0)
        v += conv_w[e * 4 + k] * xz[(size_t)(s * 256 + tt) * 2560 + e];
    }
    v = v * sigf(v);  // silu
  }
  xs[(size_t)row * 1280 + e] = v;
  xs_bf[(size_t)row * 1280 + e] = f2bf(v);
}

// ---------------------------------------------------------------------------
// Selective scan: one workgroup per segment; state (1280x16) in registers
// (5 channels x 16 states per lane). Writes bf16 y*silu(z) for out_proj GEMM.
// ---------------------------------------------------------------------------
__global__ __launch_bounds__(256) void scan_kernel(
    const float* __restrict__ draw, const float* __restrict__ dt_b,
    const float* __restrict__ A_log, const float* __restrict__ Dp,
    const float* __restrict__ xs, const float* __restrict__ dbc,
    const float* __restrict__ xz, const long long* __restrict__ seg,
    __bf16* __restrict__ yg_bf)
{
  int s = blockIdx.x, tid = threadIdx.x;
  int L = (int)(seg[2 * s + 1] - seg[2 * s] + 1);
  __shared__ float BC[32];

  float hreg[5][16];
  float areg[5][16];
  float dtb[5], dcoef[5];
#pragma unroll
  for (int j = 0; j < 5; ++j) {
    int e = tid + 256 * j;
    dtb[j] = dt_b[e];
    dcoef[j] = Dp[e];
#pragma unroll
    for (int n = 0; n < 16; ++n) {
      areg[j][n] = -__expf(A_log[(size_t)e * 16 + n]);
      hreg[j][n] = 0.f;
    }
  }

  for (int t = 0; t < L; ++t) {
    size_t row = (size_t)(s * 256 + t);
    if (tid < 32) BC[tid] = dbc[row * 80 + 40 + tid];
    __syncthreads();

#pragma unroll
    for (int j = 0; j < 5; ++j) {
      int e = tid + 256 * j;
      float xse = xs[row * 1280 + e];
      float dr = draw[row * 1280 + e] + dtb[j];
      float delta = (dr > 20.f) ? dr : log1pf(__expf(dr));
      float z = xz[row * 2560 + 1280 + e];
      float y = dcoef[j] * xse;
#pragma unroll
      for (int n = 0; n < 16; ++n) {
        float dA = __expf(delta * areg[j][n]);
        hreg[j][n] = dA * hreg[j][n] + delta * BC[n] * xse;
        y += hreg[j][n] * BC[16 + n];
      }
      yg_bf[row * 1280 + e] = f2bf(y * (z * sigf(z)));
    }
    __syncthreads();
  }
}

// ---------------------------------------------------------------------------
// Attention pooling + LayerNorm + classifier head: one workgroup per segment
// ---------------------------------------------------------------------------
__device__ __forceinline__ float blk_sum(float v, float* red) {
  int tid = threadIdx.x;
  red[tid] = v; __syncthreads();
  for (int o = 128; o > 0; o >>= 1) {
    if (tid < o) red[tid] += red[tid + o];
    __syncthreads();
  }
  float r = red[0]; __syncthreads();
  return r;
}

__global__ __launch_bounds__(256) void head_kernel(
    const float* __restrict__ te2, const long long* __restrict__ seg,
    const float* __restrict__ att_w, const float* __restrict__ att_b,
    const float* __restrict__ ln_w, const float* __restrict__ ln_b,
    const float* __restrict__ cls_w, const float* __restrict__ cls_b,
    float* __restrict__ out)
{
  int s = blockIdx.x, tid = threadIdx.x;
  int L = (int)(seg[2 * s + 1] - seg[2 * s] + 1);
  __shared__ float red[256];
  __shared__ float att[256];
  __shared__ float summ[640];

  // relu(att logits)
  float rv = -1e30f;
  if (tid < L) {
    const float* rowp = te2 + (size_t)(s * 256 + tid) * 640;
    float r = 0.f;
    for (int c = 0; c < 640; ++c) r += att_w[c] * rowp[c];
    r += att_b[0];
    rv = r > 0.f ? r : 0.f;
  }
  att[tid] = rv;
  __syncthreads();

  // softmax over t
  red[tid] = rv; __syncthreads();
  for (int o = 128; o > 0; o >>= 1) {
    if (tid < o) red[tid] = fmaxf(red[tid], red[tid + o]);
    __syncthreads();
  }
  float m = red[0]; __syncthreads();
  float e = (tid < L) ? __expf(att[tid] - m) : 0.f;
  float denom = blk_sum(e, red);
  att[tid] = e / denom;
  __syncthreads();

  // weighted sum / sqrt(L)
  float inv = rsqrtf((float)L);
  for (int c = tid; c < 640; c += 256) {
    float acc = 0.f;
    for (int t = 0; t < L; ++t)
      acc += te2[(size_t)(s * 256 + t) * 640 + c] * att[t];
    summ[c] = acc * inv;
  }
  __syncthreads();

  // layernorm + relu + classifier
  float p = 0.f;
  for (int c = tid; c < 640; c += 256) p += summ[c];
  float mu = blk_sum(p, red) / 640.f;
  p = 0.f;
  for (int c = tid; c < 640; c += 256) { float d = summ[c] - mu; p += d * d; }
  float var = blk_sum(p, red) / 640.f;
  float rstd = rsqrtf(var + 1e-5f);
  p = 0.f;
  for (int c = tid; c < 640; c += 256) {
    float sv = (summ[c] - mu) * rstd * ln_w[c] + ln_b[c];
    sv = sv > 0.f ? sv : 0.f;
    p += cls_w[c] * sv;
  }
  float dot = blk_sum(p, red);
  if (tid == 0) out[s] = sigf(dot + cls_b[0]);
}

// ---------------------------------------------------------------------------
// Host orchestration
// ---------------------------------------------------------------------------
static inline size_t align256(size_t x) { return (x + 255) & ~(size_t)255; }

extern "C" void kernel_launch(void* const* d_in, const int* in_sizes, int n_in,
                              void* d_out, int out_size, void* d_ws, size_t ws_size,
                              hipStream_t stream) {
  (void)in_sizes; (void)n_in; (void)out_size; (void)ws_size;

  const float* emb    = (const float*)d_in[0];
  const float* Wih_f  = (const float*)d_in[1];
  const float* Whh_f  = (const float*)d_in[2];
  const float* b_f    = (const float*)d_in[3];
  const float* Wih_b  = (const float*)d_in[4];
  const float* Whh_b  = (const float*)d_in[5];
  const float* b_b    = (const float*)d_in[6];
  const float* att_w  = (const float*)d_in[7];
  const float* att_b  = (const float*)d_in[8];
  const float* ln_w   = (const float*)d_in[9];
  const float* ln_b   = (const float*)d_in[10];
  const float* cls_w  = (const float*)d_in[11];
  const float* cls_b  = (const float*)d_in[12];
  const float* mnorm  = (const float*)d_in[13];
  const float* mip    = (const float*)d_in[14];
  const float* mcw    = (const float*)d_in[15];
  const float* mcb    = (const float*)d_in[16];
  const float* mxp    = (const float*)d_in[17];
  const float* mdtw   = (const float*)d_in[18];
  const float* mdtb   = (const float*)d_in[19];
  const float* malog  = (const float*)d_in[20];
  const float* mD     = (const float*)d_in[21];
  const float* mop    = (const float*)d_in[22];
  const long long* seg = (const long long*)d_in[23];
  float* out = (float*)d_out;

  const size_t ROWS = 2048;  // 8 segments x 256 padded rows
  char* ws = (char*)d_ws;
  size_t o = 0;
  auto carve = [&](size_t bytes) { size_t r = o; o = align256(o + bytes); return r; };

  __bf16* wihf_bf = (__bf16*)(ws + carve(1280ull * 1280 * 2));
  __bf16* wihb_bf = (__bf16*)(ws + carve(1280ull * 1280 * 2));
  __bf16* ip_bf   = (__bf16*)(ws + carve(2560ull * 640 * 2));
  __bf16* xp_bf   = (__bf16*)(ws + carve(72ull * 1280 * 2));
  __bf16* dtw_bf  = (__bf16*)(ws + carve(1280ull * 64 * 2));
  __bf16* op_bf   = (__bf16*)(ws + carve(640ull * 1280 * 2));
  __bf16* bufA    = (__bf16*)(ws + carve(ROWS * 1280 * 2));
  __bf16* dt_bf   = (__bf16*)(ws + carve(ROWS * 64 * 2));
  float* pre_f    = (float*)(ws + carve(ROWS * 1280 * 4));
  float* pre_b    = (float*)(ws + carve(ROWS * 1280 * 4));
  float* te       = (float*)(ws + carve(ROWS * 640 * 4));
  float* xcur     = (float*)(ws + carve(ROWS * 640 * 4));
  float* xz       = (float*)(ws + carve(ROWS * 2560 * 4));
  float* xs       = (float*)(ws + carve(ROWS * 1280 * 4));
  float* dbc      = (float*)(ws + carve(ROWS * 80 * 4));
  float* draw     = (float*)(ws + carve(ROWS * 1280 * 4));
  float* mout     = (float*)(ws + carve(ROWS * 640 * 4));

  dim3 b256(256);

  // ---- BiLSTM input GEMMs ----
  cvt_bf16_kernel<<<(1280 * 1280 + 255) / 256, b256, 0, stream>>>(Wih_f, wihf_bf, 1280 * 1280);
  cvt_bf16_kernel<<<(1280 * 1280 + 255) / 256, b256, 0, stream>>>(Wih_b, wihb_bf, 1280 * 1280);
  gather_emb_kernel<<<dim3(ROWS, 5), b256, 0, stream>>>(emb, seg, bufA);

  gemm_bf16_kernel<<<dim3(ROWS / 64, 1280 / 64), dim3(128), 0, stream>>>(
      bufA, wihf_bf, pre_f, (int)ROWS, 1280, 1280, 1280);
  gemm_bf16_kernel<<<dim3(ROWS / 64, 1280 / 64), dim3(128), 0, stream>>>(
      bufA, wihb_bf, pre_b, (int)ROWS, 1280, 1280, 1280);

  lstm_kernel<<<16, b256, 0, stream>>>(pre_f, pre_b, Whh_f, Whh_b, b_f, b_b, seg, te, xcur);

  // ---- 12 Mamba layers ----
  for (int l = 0; l < 12; ++l) {
    const float* norm_l = mnorm + (size_t)l * 640;
    const float* ip_l   = mip   + (size_t)l * 2560 * 640;
    const float* cw_l   = mcw   + (size_t)l * 1280 * 4;
    const float* cb_l   = mcb   + (size_t)l * 1280;
    const float* xp_l   = mxp   + (size_t)l * 72 * 1280;
    const float* dtw_l  = mdtw  + (size_t)l * 1280 * 40;
    const float* dtb_l  = mdtb  + (size_t)l * 1280;
    const float* alog_l = malog + (size_t)l * 1280 * 16;
    const float* D_l    = mD    + (size_t)l * 1280;
    const float* op_l   = mop   + (size_t)l * 640 * 1280;

    cvt_bf16_kernel<<<(2560 * 640 + 255) / 256, b256, 0, stream>>>(ip_l, ip_bf, 2560 * 640);
    cvt_bf16_kernel<<<(72 * 1280 + 255) / 256, b256, 0, stream>>>(xp_l, xp_bf, 72 * 1280);
    cvt_dtw_kernel<<<(1280 * 64 + 255) / 256, b256, 0, stream>>>(dtw_l, dtw_bf);
    cvt_bf16_kernel<<<(640 * 1280 + 255) / 256, b256, 0, stream>>>(op_l, op_bf, 640 * 1280);

    rmsnorm_bf16_kernel<<<ROWS, b256, 0, stream>>>(xcur, norm_l, bufA);

    // xz = xn @ in_proj^T   [2048 x 2560]
    gemm_bf16_kernel<<<dim3(ROWS / 64, 2560 / 64), dim3(128), 0, stream>>>(
        bufA, ip_bf, xz, (int)ROWS, 2560, 640, 2560);

    conv_silu_kernel<<<dim3(ROWS, 5), b256, 0, stream>>>(xz, cw_l, cb_l, seg, xs, bufA);

    // dbc = xs @ x_proj^T   [2048 x 72] (ldc 80)
    gemm_bf16_kernel<<<dim3(ROWS / 64, 2), dim3(128), 0, stream>>>(
        bufA, xp_bf, dbc, (int)ROWS, 72, 1280, 80);

    cvt_dt_kernel<<<(2048 * 64 + 255) / 256, b256, 0, stream>>>(dbc, dt_bf);

    // draw = dt @ dt_w^T    [2048 x 1280] (K padded 40->64)
    gemm_bf16_kernel<<<dim3(ROWS / 64, 1280 / 64), dim3(128), 0, stream>>>(
        dt_bf, dtw_bf, draw, (int)ROWS, 1280, 64, 1280);

    scan_kernel<<<8, b256, 0, stream>>>(draw, dtb_l, alog_l, D_l, xs, dbc, xz, seg, bufA);

    // mout = yg @ out_proj^T [2048 x 640]
    gemm_bf16_kernel<<<dim3(ROWS / 64, 640 / 64), dim3(128), 0, stream>>>(
        bufA, op_bf, mout, (int)ROWS, 640, 1280, 640);

    add_kernel<<<(int)(ROWS * 640 / 256), b256, 0, stream>>>(xcur, mout, (int)(ROWS * 640));
  }

  // te2 = mamba(te) + te
  add_kernel<<<(int)(ROWS * 640 / 256), b256, 0, stream>>>(xcur, te, (int)(ROWS * 640));

  head_kernel<<<8, b256, 0, stream>>>(xcur, seg, att_w, att_b, ln_w, ln_b, cls_w, cls_b, out);
}
